// MultiHeadAttention_7267084665455
// MI455X (gfx1250) — matrix operand order repaired and checked by reading, not compiled, not run on hardware
//
#include <hip/hip_runtime.h>
#include <stdint.h>

// ---------------------------------------------------------------------------
// MHA for MI455X (gfx1250): bf16 WMMA (16x16x32, f32 accum), flash attention,
// async global->LDS staging (double-buffered K/V), mask folded into WMMA C.
// ---------------------------------------------------------------------------

typedef __attribute__((ext_vector_type(16))) __bf16 v16bf;
typedef __attribute__((ext_vector_type(8)))  float  v8f;

union FragB { v16bf v; uint32_t u[8]; };
static_assert(sizeof(FragB) == 32, "frag size");

constexpr int   BATCH = 4;
constexpr int   SEQ   = 2048;             // T
constexpr int   HEADS = 16;
constexpr int   HDIM  = 64;               // key size
constexpr int   MS    = 1024;             // model size
constexpr int   MROWS = BATCH * SEQ;      // 8192
constexpr float DENOM = 0.125f;           // 1/sqrt(64)
constexpr float NEGINF = -1e30f;

#define DEVFN static __device__ __forceinline__

DEVFN uint16_t f2bf(float f) {
  uint32_t u = __float_as_uint(f);
  u += 0x7FFFu + ((u >> 16) & 1u);        // round-to-nearest-even
  return (uint16_t)(u >> 16);
}

DEVFN v8f zero8() { v8f z = {0.f,0.f,0.f,0.f,0.f,0.f,0.f,0.f}; return z; }

// A-fragment (16x32 bf16) per-lane K offset for VGPR g:
//   lanes 0-15 rows M=0..15 hold K {0..7,16..23}; lanes 16-31 hold K {8..15,24..31}
DEVFN int a_kk(int g, int khalf) { return khalf * 8 + ((g & 4) << 2) + ((g & 3) << 1); }
// B-fragment (32x16 bf16): same K order as A: lanes 0-15 hold K {0..7,16..23}; lanes 16-31 hold K {8..15,24..31}
DEVFN int b_kk(int g, int khalf) { return khalf * 8 + ((g & 4) << 2) + ((g & 3) << 1); }

// LDS byte offset of a __shared__ pointer: low 32 bits of the flat address
// (LDS aperture: addr[31:0] is the wave-relative LDS offset, ISA 10.2).
DEVFN uint32_t lds_off(const void* p) { return (uint32_t)(uint64_t)(uintptr_t)p; }

// Async global -> LDS copy of 16 bytes (per lane), tracked by ASYNCcnt.
DEVFN void async_copy_b128(uint32_t lds_byte_off, const void* gptr) {
  const uint64_t ga = (uint64_t)(uintptr_t)gptr;
  asm volatile("global_load_async_to_lds_b128 %0, %1, off"
               :: "v"(lds_byte_off), "v"(ga) : "memory");
}
DEVFN void wait_async0() {
  asm volatile("s_wait_asynccnt 0x0" ::: "memory");
}

// ---------------------------------------------------------------------------
// GEMM: C[M,N] = A[M,K] * W[K,N] + bias, M=8192, K=N=1024.
//   A_IS_F32:  A is fp32 (converted to bf16 while staging to LDS);
//              else bf16, staged with async global->LDS copies.
//   OUT_MODE:  0 = fp32 row-major [M,N]
//              1 = bf16 [B,H,T,64]  ((acc+bias)*scale)
//              2 = bf16 [B,H,64,T]  (transposed head layout, for V)
// Block: 256 thr (8 waves), 128x64 C tile, 8 waves as 4(m) x 2(n) of 32x32.
// ---------------------------------------------------------------------------
template<bool A_IS_F32, int OUT_MODE>
__global__ __launch_bounds__(256)
void gemm_kernel(const void* __restrict__ Aptr, const float* __restrict__ Wp,
                 const float* __restrict__ bias, void* __restrict__ Cptr,
                 float scale)
{
  constexpr int K   = MS;
  constexpr int N   = MS;
  constexpr int LDA = 40;                       // padded LDS stride (elems, even)
  __shared__ uint16_t lA [128 * LDA];           // A tile  [128][32]
  __shared__ uint16_t lBt[64  * LDA];           // W tile transposed: [n][k] (32x64)

  const int tid   = threadIdx.x;
  const int lane  = tid & 31;
  const int wave  = tid >> 5;
  const int wm    = (wave & 3) * 32;
  const int wn    = (wave >> 2) * 32;
  const int n0    = blockIdx.x * 64;
  const int m0    = blockIdx.y * 128;
  const int lrow  = lane & 15;
  const int khalf = lane >> 4;

  v8f acc[2][2];
  for (int i = 0; i < 2; ++i) for (int j = 0; j < 2; ++j) acc[i][j] = zero8();

  for (int k0 = 0; k0 < K; k0 += 32) {
    // stage A (128x32)
    if constexpr (A_IS_F32) {
      for (int idx = tid; idx < 128 * 32; idx += 256) {
        const int r = idx >> 5, c = idx & 31;
        lA[r * LDA + c] = f2bf(((const float*)Aptr)[(size_t)(m0 + r) * K + k0 + c]);
      }
    } else {
      // async copy: 128 rows x 4 chunks of 8 bf16 (16B), 512 chunks over 256 thr
      #pragma unroll
      for (int it = 0; it < 2; ++it) {
        const int chunk = tid + it * 256;
        const int r = chunk >> 2, c8 = (chunk & 3) << 3;
        async_copy_b128(lds_off(&lA[r * LDA + c8]),
                        (const uint16_t*)Aptr + (size_t)(m0 + r) * K + k0 + c8);
      }
    }
    // stage W (32x64) transposed so B-frag K-pairs are contiguous 4B words
    for (int idx = tid; idx < 32 * 64; idx += 256) {
      const int r = idx >> 6, c = idx & 63;
      lBt[c * LDA + r] = f2bf(Wp[(size_t)(k0 + r) * N + n0 + c]);
    }
    if constexpr (!A_IS_F32) wait_async0();
    __syncthreads();

    FragB af[2], bf2[2];
    #pragma unroll
    for (int mi = 0; mi < 2; ++mi) {
      const int mrow = wm + mi * 16 + lrow;
      #pragma unroll
      for (int g = 0; g < 8; ++g)
        af[mi].u[g] = *(const uint32_t*)&lA[mrow * LDA + a_kk(g, khalf)];
    }
    #pragma unroll
    for (int ni = 0; ni < 2; ++ni) {
      const int ncol = wn + ni * 16 + lrow;
      #pragma unroll
      for (int g = 0; g < 8; ++g)
        bf2[ni].u[g] = *(const uint32_t*)&lBt[ncol * LDA + b_kk(g, khalf)];
    }
    #pragma unroll
    for (int mi = 0; mi < 2; ++mi)
      #pragma unroll
      for (int ni = 0; ni < 2; ++ni)
        acc[mi][ni] = __builtin_amdgcn_wmma_f32_16x16x32_bf16(
            false, af[mi].v, false, bf2[ni].v, (short)0, acc[mi][ni], false, false);
    __syncthreads();
  }

  // epilogue: C layout lane mapping  M = r + 8*khalf, N = lrow
  #pragma unroll
  for (int mi = 0; mi < 2; ++mi) {
    #pragma unroll
    for (int ni = 0; ni < 2; ++ni) {
      #pragma unroll
      for (int r = 0; r < 8; ++r) {
        const int m = m0 + wm + mi * 16 + r + khalf * 8;
        const int n = n0 + wn + ni * 16 + lrow;
        float v = acc[mi][ni][r] + bias[n];
        if constexpr (OUT_MODE == 0) {
          ((float*)Cptr)[(size_t)m * N + n] = v;
        } else {
          v *= scale;
          const int b = m >> 11, t = m & (SEQ - 1);   // SEQ = 2048
          const int h = n >> 6,  d = n & 63;
          if constexpr (OUT_MODE == 1)
            ((uint16_t*)Cptr)[(((size_t)(b * HEADS + h) * SEQ + t) << 6) + d] = f2bf(v);
          else // OUT_MODE == 2: [B,H,D,T] for async V staging in attention
            ((uint16_t*)Cptr)[((size_t)(b * HEADS + h) * HDIM + d) * SEQ + t] = f2bf(v);
        }
      }
    }
  }
}

// ---------------------------------------------------------------------------
// Flash attention: one block per (128 q-rows, head, batch).
// Q/K bf16 [B,H,T,64]; V bf16 transposed [B,H,64,T]. K/V blocks staged with
// double-buffered async global->LDS copies. Mask folded into WMMA C-init.
// Online softmax in registers (shfl reductions over 16-lane N-groups).
// Output bf16 [B,T,MS] for the O-projection.
// ---------------------------------------------------------------------------
__global__ __launch_bounds__(256)
void attn_kernel(const uint16_t* __restrict__ Qb, const uint16_t* __restrict__ Kb,
                 const uint16_t* __restrict__ Vtb, const int* __restrict__ maskp,
                 uint16_t* __restrict__ Ob)
{
  constexpr int LQ = 72;                        // padded stride (even; 16B-aligned rows)
  __shared__ uint16_t lK [2][64 * LQ];          // K block  [key][dim]
  __shared__ uint16_t lVt[2][64 * LQ];          // V block  [dim][key] (already transposed)
  __shared__ uint16_t lP [128 * LQ];            // softmax weights bf16 [qrow][key]

  const int tid   = threadIdx.x;
  const int lane  = tid & 31;
  const int wave  = tid >> 5;
  const int lrow  = lane & 15;
  const int khalf = lane >> 4;
  const int qb0   = blockIdx.x * 128;
  const int h     = blockIdx.y;
  const int b     = blockIdx.z;

  const size_t headbase = ((size_t)(b * HEADS + h) * SEQ) << 6;   // [B,H,T,64]
  const size_t vthead   = (size_t)(b * HEADS + h) * HDIM * SEQ;   // [B,H,64,T]

  // issue async copies for one 64-key block into buffer `buf`
  auto stage_kv = [&](int buf, int kb) {
    #pragma unroll
    for (int it = 0; it < 2; ++it) {
      const int chunk = tid + it * 256;          // 512 chunks of 8 bf16 (16B)
      const int r  = chunk >> 3;                 // 64 rows
      const int c8 = (chunk & 7) << 3;           // 8 chunks per 64-elem row
      async_copy_b128(lds_off(&lK[buf][r * LQ + c8]),
                      Kb + headbase + ((size_t)(kb + r) << 6) + c8);
      async_copy_b128(lds_off(&lVt[buf][r * LQ + c8]),
                      Vtb + vthead + (size_t)r * SEQ + kb + c8);
    }
  };

  // Q fragments straight from global (rows contiguous, K-pairs are 4B words)
  FragB aq[2];
  {
    const int qrow = qb0 + wave * 16 + lrow;
    const uint16_t* qp = Qb + headbase + ((size_t)qrow << 6);
    #pragma unroll
    for (int ks = 0; ks < 2; ++ks)
      #pragma unroll
      for (int g = 0; g < 8; ++g)
        aq[ks].u[g] = *(const uint32_t*)&qp[ks * 32 + a_kk(g, khalf)];
  }

  v8f o[4];
  for (int i = 0; i < 4; ++i) o[i] = zero8();
  float m_run[8], l_run[8];
  #pragma unroll
  for (int r = 0; r < 8; ++r) { m_run[r] = NEGINF; l_run[r] = 0.f; }

  stage_kv(0, 0);

  for (int kb0 = 0; kb0 < SEQ; kb0 += 64) {
    const int buf = (kb0 >> 6) & 1;
    wait_async0();
    __syncthreads();
    if (kb0 + 64 < SEQ) {
      stage_kv(buf ^ 1, kb0 + 64);
      // prefetch next mask block (128 rows x 64 cols of int32)
      const int qr = qb0 + (tid >> 1);
      const int cc = (tid & 1) << 5;
      __builtin_prefetch(&maskp[((size_t)b * SEQ + qr) * SEQ + kb0 + 64 + cc], 0, 1);
    }

    // mask folded into the WMMA C operand: -1e30 + logit == -1e30 in fp32
    v8f s[4];
    #pragma unroll
    for (int nt = 0; nt < 4; ++nt) {
      #pragma unroll
      for (int r = 0; r < 8; ++r) {
        const int qrow = qb0 + wave * 16 + r + khalf * 8;
        const int col  = kb0 + nt * 16 + lrow;
        s[nt][r] = maskp[((size_t)b * SEQ + qrow) * SEQ + col] ? 0.f : NEGINF;
      }
    }

    // logits tile: 16 q-rows x 64 keys per wave (4 n-tiles x 2 k-steps)
    #pragma unroll
    for (int ks = 0; ks < 2; ++ks) {
      #pragma unroll
      for (int nt = 0; nt < 4; ++nt) {
        FragB bk;
        const int ncol = nt * 16 + lrow;
        #pragma unroll
        for (int g = 0; g < 8; ++g)
          bk.u[g] = *(const uint32_t*)&lK[buf][ncol * LQ + ks * 32 + b_kk(g, khalf)];
        s[nt] = __builtin_amdgcn_wmma_f32_16x16x32_bf16(
            false, aq[ks].v, false, bk.v, (short)0, s[nt], false, false);
      }
    }

    // online softmax, fully in registers; row spans 16 lanes x 4 regs
    float scl[8];
    #pragma unroll
    for (int r = 0; r < 8; ++r) {
      float mx = fmaxf(fmaxf(s[0][r], s[1][r]), fmaxf(s[2][r], s[3][r]));
      #pragma unroll
      for (int off = 1; off < 16; off <<= 1)
        mx = fmaxf(mx, __shfl_xor(mx, off, 32));
      const float mnew = fmaxf(m_run[r], mx);
      const float sc   = __expf(m_run[r] - mnew);
      float rsum = 0.f;
      #pragma unroll
      for (int nt = 0; nt < 4; ++nt) {
        const float p = __expf(s[nt][r] - mnew);
        s[nt][r] = p;
        rsum += p;
      }
      #pragma unroll
      for (int off = 1; off < 16; off <<= 1)
        rsum += __shfl_xor(rsum, off, 32);
      l_run[r] = l_run[r] * sc + rsum;
      m_run[r] = mnew;
      scl[r]   = sc;
    }

    // store P (C-layout regs -> row-major bf16 in LDS) and rescale O
    #pragma unroll
    for (int nt = 0; nt < 4; ++nt)
      #pragma unroll
      for (int r = 0; r < 8; ++r)
        lP[(wave * 16 + r + khalf * 8) * LQ + nt * 16 + lrow] = f2bf(s[nt][r]);
    #pragma unroll
    for (int dt = 0; dt < 4; ++dt)
      #pragma unroll
      for (int r = 0; r < 8; ++r)
        o[dt][r] *= scl[r];
    __syncthreads();

    // PV: A = P (16x64), B = V block (64x16 per d-tile)
    FragB ap[2];
    #pragma unroll
    for (int ks = 0; ks < 2; ++ks) {
      const int row = wave * 16 + lrow;
      #pragma unroll
      for (int g = 0; g < 8; ++g)
        ap[ks].u[g] = *(const uint32_t*)&lP[row * LQ + ks * 32 + a_kk(g, khalf)];
    }
    #pragma unroll
    for (int ks = 0; ks < 2; ++ks) {
      #pragma unroll
      for (int dt = 0; dt < 4; ++dt) {
        FragB bv;
        const int dcol = dt * 16 + lrow;
        #pragma unroll
        for (int g = 0; g < 8; ++g)
          bv.u[g] = *(const uint32_t*)&lVt[buf][dcol * LQ + ks * 32 + b_kk(g, khalf)];
        o[dt] = __builtin_amdgcn_wmma_f32_16x16x32_bf16(
            false, ap[ks].v, false, bv.v, (short)0, o[dt], false, false);
      }
    }
    // no trailing barrier: next iteration begins with wait_async0+barrier,
    // and lP / lK / lVt are only rewritten after that barrier.
  }

  // normalize and store bf16 in [B,T,MS]
  #pragma unroll
  for (int dt = 0; dt < 4; ++dt) {
    #pragma unroll
    for (int r = 0; r < 8; ++r) {
      const int   trow = qb0 + wave * 16 + r + khalf * 8;
      const int   dcol = h * 64 + dt * 16 + lrow;
      const float val  = o[dt][r] / l_run[r];
      Ob[(size_t)(b * SEQ + trow) * MS + dcol] = f2bf(val);
    }
  }
}

// ---------------------------------------------------------------------------
// Launch: 3x QKV projections -> flash attention -> O projection.
// Workspace: Q,K bf16 [B,H,T,64], V bf16 [B,H,64,T], attn bf16 [B,T,MS]
//            = 4 x 16 MB = 67 MB.
// ---------------------------------------------------------------------------
extern "C" void kernel_launch(void* const* d_in, const int* in_sizes, int n_in,
                              void* d_out, int out_size, void* d_ws, size_t ws_size,
                              hipStream_t stream)
{
  (void)in_sizes; (void)n_in; (void)out_size; (void)ws_size;

  const float* query = (const float*)d_in[0];
  const float* key   = (const float*)d_in[1];
  const float* value = (const float*)d_in[2];
  const int*   mask  = (const int*)  d_in[3];
  const float* Wq    = (const float*)d_in[4];
  const float* bq    = (const float*)d_in[5];
  const float* Wk    = (const float*)d_in[6];
  const float* bk    = (const float*)d_in[7];
  const float* Wv    = (const float*)d_in[8];
  const float* bv    = (const float*)d_in[9];
  const float* Wo    = (const float*)d_in[10];
  const float* bo    = (const float*)d_in[11];

  constexpr size_t QKV_ELEMS = (size_t)BATCH * HEADS * SEQ * HDIM;  // 8388608
  uint16_t* Qb  = (uint16_t*)d_ws;
  uint16_t* Kb  = Qb + QKV_ELEMS;
  uint16_t* Vtb = Kb + QKV_ELEMS;       // [B, H, 64, T]
  uint16_t* Ab  = Vtb + QKV_ELEMS;      // [B, T, MS]

  const dim3 blk(256);
  const dim3 gproj(MS / 64, MROWS / 128);          // (16, 64)
  const dim3 gattn(SEQ / 128, HEADS, BATCH);       // (16, 16, 4)

  gemm_kernel<true,  1><<<gproj, blk, 0, stream>>>(query, Wq, bq, Qb,  DENOM);
  gemm_kernel<true,  1><<<gproj, blk, 0, stream>>>(key,   Wk, bk, Kb,  1.0f);
  gemm_kernel<true,  2><<<gproj, blk, 0, stream>>>(value, Wv, bv, Vtb, 1.0f);
  attn_kernel<<<gattn, blk, 0, stream>>>(Qb, Kb, Vtb, mask, Ab);
  gemm_kernel<false, 0><<<gproj, blk, 0, stream>>>(Ab, Wo, bo, d_out, 1.0f);
}
